// MNERHeadBiLSTM_249108103791
// MI455X (gfx1250) — matrix operand order repaired and checked by reading, not compiled, not run on hardware
//
#include <hip/hip_runtime.h>
#include <hip/hip_bf16.h>

// ---------------- CDNA5 WMMA vector types ----------------
typedef __attribute__((ext_vector_type(16))) __bf16 v16bf;
typedef __attribute__((ext_vector_type(8)))  __bf16 v8bf;
typedef __attribute__((ext_vector_type(8)))  float  v8f;
typedef unsigned int u32x4 __attribute__((ext_vector_type(4)));
typedef int          i32x4 __attribute__((ext_vector_type(4)));
typedef int          i32x8 __attribute__((ext_vector_type(8)));

static constexpr int kB  = 64;
static constexpr int kT  = 512;
static constexpr int kD  = 768;
static constexpr int kH  = 256;
static constexpr int kL  = 9;
static constexpr int k4H = 1024;
static constexpr int kM  = kB * kT;   // 32768 token rows

// ---------------- feature detection (device pass only) ----------------
#if defined(__AMDGCN__) && __has_builtin(__builtin_amdgcn_tensor_load_to_lds)
#define HAVE_TDM 1
#else
#define HAVE_TDM 0
#endif
#if defined(__AMDGCN__) && __has_builtin(__builtin_amdgcn_global_load_async_to_lds_b128)
#define HAVE_ASYNC 1
#else
#define HAVE_ASYNC 0
#endif
#if defined(__AMDGCN__) && __has_builtin(__builtin_amdgcn_s_wait_asynccnt)
#define HAVE_WAIT_ASYNC 1
#else
#define HAVE_WAIT_ASYNC 0
#endif
#if defined(__AMDGCN__) && __has_builtin(__builtin_amdgcn_s_wait_tensorcnt)
#define HAVE_WAIT_TENSOR 1
#else
#define HAVE_WAIT_TENSOR 0
#endif

// ---------------- WMMA helpers ----------------
__device__ __forceinline__ v8f wmma_bf16(v16bf a, v16bf b, v8f c) {
  // D = A(16x32) x B(32x16) + C, f32 accumulate
  return __builtin_amdgcn_wmma_f32_16x16x32_bf16(false, a, false, b, (short)0, c,
                                                 false, false);
}

// A-matrix 16x32 bf16 fragment: lane (half h) holds K in [k0+8h, k0+8h+8) and
// [k0+16+8h, k0+16+8h+8) per the CDNA5 ISA VGPR layout (two 16B loads).
__device__ __forceinline__ v16bf load_a16(const __bf16* row_k0, int half) {
  v8bf lo = *(const v8bf*)(row_k0 + 8 * half);
  v8bf hi = *(const v8bf*)(row_k0 + 16 + 8 * half);
  v16bf r;
#pragma unroll
  for (int j = 0; j < 8; ++j) { r[j] = lo[j]; r[j + 8] = hi[j]; }
  return r;
}

// B-matrix 32x16 bf16 fragment: lane (col = lane%16, half h) holds K in
// [k0+8h, k0+8h+8) and [k0+16+8h, k0+16+8h+8) (two 16B loads). row_k0 points at W[n, k0].
__device__ __forceinline__ v16bf load_b16(const __bf16* row_k0, int half) {
  return load_a16(row_k0, half);
}

__device__ __forceinline__ float sigm(float x) {
  return 1.0f / (1.0f + __expf(-x));
}

// ---------------- CDNA5 data movers ----------------
// 16B global -> LDS copy. Async path uses GLOBAL_LOAD_ASYNC_TO_LDS_B128
// (ASYNCcnt); fallback is a synchronous register round-trip.
__device__ __forceinline__ void cp16_g2l(__bf16* l, const __bf16* g) {
#if HAVE_ASYNC
  __bf16* gnc = (__bf16*)g;  // builtin takes non-const pointers
  __builtin_amdgcn_global_load_async_to_lds_b128(
      (__attribute__((address_space(1))) i32x4*)gnc,
      (__attribute__((address_space(3))) i32x4*)l, 0, 0);
#else
  *(v8bf*)l = *(const v8bf*)g;
#endif
}

__device__ __forceinline__ void wait_async0() {
#if HAVE_ASYNC
#if HAVE_WAIT_ASYNC
  __builtin_amdgcn_s_wait_asynccnt(0);
#else
  asm volatile("s_wait_asynccnt 0x0" ::: "memory");
#endif
#endif
}

#if HAVE_TDM
__device__ __forceinline__ unsigned lds_offset(const void* p) {
  return (unsigned)(unsigned long long)(__attribute__((address_space(3))) const char*)p;
}

// 2D bf16 tile load via the Tensor Data Mover. Tile = rows x rowElems
// elements, row stride = strideElems (elements). D# per CDNA5 ISA ch.8.
__device__ __forceinline__ void tdm_load_2d_bf16(unsigned lds_off,
                                                 unsigned long long gaddr,
                                                 int rows, int rowElems,
                                                 int strideElems) {
  u32x4 g0;
  g0[0] = 1u;  // count=1 (valid), is_restore=0, gather_mode=0
  g0[1] = lds_off;                                   // lds_addr
  g0[2] = (unsigned)(gaddr & 0xFFFFFFFFull);         // global_addr[31:0]
  g0[3] = (unsigned)((gaddr >> 32) & 0x1FFFFFFull)   // global_addr[56:32]
          | (2u << 30);                              // type=2 ("image")
  i32x8 g1;
  g1[0] = (1 << 16);                                  // data_size=1 (2B), mask=0
  g1[1] = (int)((unsigned)(rowElems & 0xFFFF) << 16); // tensor_dim0[15:0]
  g1[2] = (int)(((unsigned)rowElems >> 16) |
                ((unsigned)(rows & 0xFFFF) << 16));   // td0 hi | tensor_dim1 lo
  g1[3] = (int)(((unsigned)rows >> 16) |
                ((unsigned)(rowElems & 0xFFFF) << 16)); // td1 hi | tile_dim0
  g1[4] = rows & 0xFFFF;                              // tile_dim1 (tile_dim2=0)
  g1[5] = strideElems;                                // tensor_dim0_stride lo32
  g1[6] = 0;                                          // stride hi, dim1_stride lo
  g1[7] = 0;
  i32x4 z4 = {0, 0, 0, 0};
  i32x8 z8 = {0, 0, 0, 0, 0, 0, 0, 0};
  __builtin_amdgcn_tensor_load_to_lds(g0, g1, z4, z4, z8, 0);
}
#endif

__device__ __forceinline__ void wait_tensor0() {
#if HAVE_TDM
#if HAVE_WAIT_TENSOR
  __builtin_amdgcn_s_wait_tensorcnt(0);
#else
  asm volatile("s_wait_tensorcnt 0x0" ::: "memory");
#endif
#endif
}

// ---------------- dtype conversion ----------------
__global__ void f32_to_bf16_kernel(const float* __restrict__ in,
                                   __bf16* __restrict__ out, int n) {
  int i = blockIdx.x * blockDim.x + threadIdx.x;
  if (i < n) out[i] = (__bf16)in[i];
}

__global__ void pad_cls_kernel(const float* __restrict__ w,  // [9][512]
                               __bf16* __restrict__ out) {   // [16][512]
  int i = blockIdx.x * blockDim.x + threadIdx.x;
  int r = i >> 9, k = i & 511;
  out[i] = (r < kL) ? (__bf16)w[r * (2 * kH) + k] : (__bf16)0.0f;
}

// ---------------- input projection GEMM: xg = X @ Wih^T + b ----------------
// X: [kM, K]; W: [2][k4H][K]; bias: [2][k4H]; xg: [2][kT][kB][k4H]
// block = 256 threads (8 waves); workgroup tile 128(M) x 64(N).
// Per 32-wide K chunk: A tile (128x32, 8KB) staged with async-to-LDS copies,
// B tile (64x32, 4KB) staged with one TDM descriptor; double buffered so the
// next chunk's DMA overlaps this chunk's WMMAs.
__global__ __launch_bounds__(256) void proj_gemm_kernel(
    const __bf16* __restrict__ X, const __bf16* __restrict__ W,
    const float* __restrict__ bias, float* __restrict__ xg, int K) {
  const int dir = blockIdx.z;
  W    += (size_t)dir * k4H * K;
  bias += dir * k4H;
  xg   += (size_t)dir * kT * kB * k4H;

  extern __shared__ char sm[];
  const int BUFB = 12 * 1024;  // 8KB A + 4KB B per buffer

  const int tid  = threadIdx.x;
  const int wv   = tid >> 5, lane = tid & 31;
  const int lr   = lane & 15, half = lane >> 4;
  const int mloc = (wv & 1) * 64;             // wave's local M base (of 128)
  const int nloc = (wv >> 1) * 16 + lr;       // wave's local N col  (of 64)
  const int mbase = blockIdx.x * 128 + mloc;
  const int n     = blockIdx.y * 64 + nloc;

  const __bf16* Xblk = X + (size_t)(blockIdx.x * 128) * K;
  const __bf16* Wblk = W + (size_t)(blockIdx.y * 64) * K;

  auto stage = [&](int buf, int k0) {
    __bf16* aSh = (__bf16*)(sm + buf * BUFB);
    __bf16* bSh = (__bf16*)(sm + buf * BUFB + 8 * 1024);
    // A tile: 512 x 16B chunks, 2 per thread (row-major [128][32])
#pragma unroll
    for (int c = 0; c < 2; ++c) {
      const int chunk = tid * 2 + c;
      const int row = chunk >> 2, part = chunk & 3;
      cp16_g2l(aSh + row * 32 + part * 8,
               Xblk + (size_t)row * K + k0 + part * 8);
    }
#if HAVE_TDM
    // B tile: one 64-row x 32-elem bf16 TDM transfer, issued by wave 0
    if (wv == 0)
      tdm_load_2d_bf16(lds_offset(bSh),
                       (unsigned long long)(uintptr_t)(Wblk + k0),
                       /*rows=*/64, /*rowElems=*/32, /*strideElems=*/K);
#else
    {
      const int row = tid >> 2, part = tid & 3;
      cp16_g2l(bSh + row * 32 + part * 8,
               Wblk + (size_t)row * K + k0 + part * 8);
    }
#endif
  };

  v8f acc[4];
#pragma unroll
  for (int i = 0; i < 4; ++i)
#pragma unroll
    for (int r = 0; r < 8; ++r) acc[i][r] = 0.0f;

  stage(0, 0);
  const int nk = K >> 5;
  for (int kc = 0; kc < nk; ++kc) {
    const int buf = kc & 1;
    wait_async0();
    wait_tensor0();
    __syncthreads();  // staged data visible; previous compute done reading buf^1
    if (kc + 1 < nk) stage(buf ^ 1, (kc + 1) << 5);

    const __bf16* aSh = (const __bf16*)(sm + buf * BUFB);
    const __bf16* bSh = (const __bf16*)(sm + buf * BUFB + 8 * 1024);
    v16bf bf = load_b16(bSh + nloc * 32, half);
#pragma unroll
    for (int i = 0; i < 4; ++i) {
      v16bf af = load_a16(aSh + (mloc + i * 16 + lr) * 32, half);
      acc[i] = wmma_bf16(af, bf, acc[i]);
    }
  }

  const float bv = bias[n];
#pragma unroll
  for (int i = 0; i < 4; ++i) {
#pragma unroll
    for (int r = 0; r < 8; ++r) {
      const int m = mbase + i * 16 + r + half * 8;
      const int bidx = m >> 9;       // m / kT
      const int t    = m & (kT - 1); // m % kT
      xg[((size_t)t * kB + bidx) * k4H + n] = acc[i][r] + bv;
    }
  }
}

// ---------------- persistent LSTM time scan (one WGP per direction) --------
// Per step: gates = xg_t + h @ Whh^T (M=64,N=1024,K=256 via WMMA, gates to
// LDS), barrier, elementwise c/h update (c in VGPRs), h -> LDS (bf16) and to
// the layer output. 1024 threads = 32 waves; wave owns 8 16x16 tiles.
// The next step's 256KB xg block is prefetched while this step computes.
__global__ __launch_bounds__(1024) void lstm_scan_kernel(
    const float* __restrict__ xg,    // [2][kT][kB][k4H]
    const __bf16* __restrict__ Whh,  // [2][k4H][kH]
    __bf16* __restrict__ hout) {     // [kB][kT][2*kH]
  const int dir = blockIdx.x;
  const float*  xgd = xg + (size_t)dir * kT * kB * k4H;
  const __bf16* Wd  = Whh + (size_t)dir * k4H * kH;

  extern __shared__ char smem[];
  float*  gates = (float*)smem;                   // [kB][k4H]  256 KB
  __bf16* hsh   = (__bf16*)(smem + kB * k4H * 4); // [kB][kH]    32 KB

  const int tid  = threadIdx.x;
  const int wv   = tid >> 5, lane = tid & 31;
  const int lr   = lane & 15, half = lane >> 4;
  const int mt   = wv & 3;   // batch tile (16 rows)
  const int ngrp = wv >> 2;  // 8 n-tiles of 16

  const int j    = tid & 255; // hidden col owned in elementwise phase
  const int bgrp = tid >> 8;  // owns batches [bgrp*16, bgrp*16+16)

  float creg[16];
#pragma unroll
  for (int i = 0; i < 16; ++i) creg[i] = 0.0f;
#pragma unroll
  for (int i = 0; i < 16; ++i) hsh[tid * 16 + i] = (__bf16)0.0f;
  __syncthreads();

  for (int s = 0; s < kT; ++s) {
    const int t = (dir == 0) ? s : (kT - 1 - s);
    const float* xgt = xgd + (size_t)t * kB * k4H;

    // prefetch next step's xg block (global_prefetch_b8); 1024 threads x 256B
    if (s + 1 < kT) {
      const int tn = (dir == 0) ? (s + 1) : (kT - 2 - s);
      __builtin_prefetch(xgd + (size_t)tn * kB * k4H + tid * 64, 0, 3);
    }

    // ---- gate GEMM: acc = xg_t ; acc += h @ Whh^T ----
    v8f acc[8];
#pragma unroll
    for (int i = 0; i < 8; ++i) {
      const int n = (ngrp * 8 + i) * 16 + lr;
#pragma unroll
      for (int r = 0; r < 8; ++r)
        acc[i][r] = xgt[(size_t)(mt * 16 + r + half * 8) * k4H + n];
    }
    const __bf16* arow = hsh + (mt * 16 + lr) * kH;
#pragma unroll
    for (int k0 = 0; k0 < kH; k0 += 32) {
      v16bf af = load_a16(arow + k0, half);
#pragma unroll
      for (int i = 0; i < 8; ++i) {
        const int n = (ngrp * 8 + i) * 16 + lr;
        v16bf bf = load_b16(Wd + (size_t)n * kH + k0, half);
        acc[i] = wmma_bf16(af, bf, acc[i]);
      }
    }
#pragma unroll
    for (int i = 0; i < 8; ++i) {
      const int n = (ngrp * 8 + i) * 16 + lr;
#pragma unroll
      for (int r = 0; r < 8; ++r)
        gates[(mt * 16 + r + half * 8) * k4H + n] = acc[i][r];
    }
    __syncthreads();

    // ---- elementwise: c = sig(f)*c + sig(i)*tanh(g); h = sig(o)*tanh(c) ----
#pragma unroll 4
    for (int kk = 0; kk < 16; ++kk) {
      const int b = bgrp * 16 + kk;
      const float gi = gates[b * k4H + j];
      const float gf = gates[b * k4H + kH + j];
      const float gg = gates[b * k4H + 2 * kH + j];
      const float go = gates[b * k4H + 3 * kH + j];
      const float c  = sigm(gf) * creg[kk] + sigm(gi) * tanhf(gg);
      creg[kk] = c;
      const float h = sigm(go) * tanhf(c);
      hsh[b * kH + j] = (__bf16)h;
      hout[((size_t)b * kT + t) * (2 * kH) + dir * kH + j] = (__bf16)h;
    }
    __syncthreads();
  }
}

// ---------------- classifier: logits = h @ W^T + b (N padded 9->16) --------
__global__ __launch_bounds__(256) void cls_gemm_kernel(
    const __bf16* __restrict__ X,   // [kM][512]
    const __bf16* __restrict__ Wp,  // [16][512] zero-padded
    const float* __restrict__ bias, // [9]
    float* __restrict__ out) {      // [kM][9]
  const int tid  = threadIdx.x;
  const int wv   = tid >> 5, lane = tid & 31;
  const int lr   = lane & 15, half = lane >> 4;
  const int mbase = blockIdx.x * 512 + wv * 64;
  const int K = 2 * kH;

  v8f acc[4];
#pragma unroll
  for (int i = 0; i < 4; ++i)
#pragma unroll
    for (int r = 0; r < 8; ++r) acc[i][r] = 0.0f;

  const __bf16* wrow = Wp + lr * K;
  for (int k0 = 0; k0 < K; k0 += 32) {
    v16bf bf = load_b16(wrow + k0, half);
#pragma unroll
    for (int i = 0; i < 4; ++i) {
      v16bf af = load_a16(X + (size_t)(mbase + i * 16 + lr) * K + k0, half);
      acc[i] = wmma_bf16(af, bf, acc[i]);
    }
  }
  if (lr < kL) {
    const float bv = bias[lr];
#pragma unroll
    for (int i = 0; i < 4; ++i)
#pragma unroll
      for (int r = 0; r < 8; ++r)
        out[(size_t)(mbase + i * 16 + r + half * 8) * kL + lr] = acc[i][r] + bv;
  }
}

// ---------------- CRF NLL on logits[:, 1:T-1] ----------------
__global__ __launch_bounds__(64) void crf_nll_kernel(
    const float* __restrict__ logits, const int* __restrict__ labels,
    const float* __restrict__ start, const float* __restrict__ endw,
    const float* __restrict__ trans, float* __restrict__ loss) {
  __shared__ float s_start[kL], s_end[kL], s_trans[kL * kL], red[kB];
  const int tid = threadIdx.x;
  if (tid < kL) { s_start[tid] = start[tid]; s_end[tid] = endw[tid]; }
  for (int i = tid; i < kL * kL; i += 64) s_trans[i] = trans[i];
  __syncthreads();

  const int b = tid;
  const int S = kT - 2;  // 510 kept timesteps
  const float* em = logits + ((size_t)b * kT + 1) * kL;
  const int*   tg = labels + b * kT + 1;

  float alpha[kL];
  float num = s_start[tg[0]] + em[tg[0]];
#pragma unroll
  for (int l = 0; l < kL; ++l) alpha[l] = s_start[l] + em[l];

  for (int t2 = 1; t2 < S; ++t2) {
    const float* e = em + t2 * kL;
    num += e[tg[t2]] + s_trans[tg[t2 - 1] * kL + tg[t2]];
    float na[kL];
#pragma unroll
    for (int l2 = 0; l2 < kL; ++l2) {
      float mx = -3.0e38f;
#pragma unroll
      for (int l = 0; l < kL; ++l)
        mx = fmaxf(mx, alpha[l] + s_trans[l * kL + l2]);
      float sum = 0.0f;
#pragma unroll
      for (int l = 0; l < kL; ++l)
        sum += __expf(alpha[l] + s_trans[l * kL + l2] - mx);
      na[l2] = mx + __logf(sum) + e[l2];
    }
#pragma unroll
    for (int l = 0; l < kL; ++l) alpha[l] = na[l];
  }
  num += s_end[tg[S - 1]];

  float mx = -3.0e38f;
#pragma unroll
  for (int l = 0; l < kL; ++l) mx = fmaxf(mx, alpha[l] + s_end[l]);
  float sum = 0.0f;
#pragma unroll
  for (int l = 0; l < kL; ++l) sum += __expf(alpha[l] + s_end[l] - mx);
  const float logZ = mx + __logf(sum);

  red[tid] = num - logZ;
  __syncthreads();
  for (int off = 32; off > 0; off >>= 1) {
    if (tid < off) red[tid] += red[tid + off];
    __syncthreads();
  }
  if (tid == 0) loss[0] = -(red[0] / (float)kB);
}

// ---------------- launcher ----------------
extern "C" void kernel_launch(void* const* d_in, const int* in_sizes, int n_in,
                              void* d_out, int out_size, void* d_ws,
                              size_t ws_size, hipStream_t stream) {
  const float* features = (const float*)d_in[0];
  const int*   labels   = (const int*)d_in[1];
  // d_in[2] mask: all-ones, unused.
  const float* w0ih = (const float*)d_in[3];
  const float* w0hh = (const float*)d_in[4];
  const float* b0   = (const float*)d_in[5];
  const float* w1ih = (const float*)d_in[6];
  const float* w1hh = (const float*)d_in[7];
  const float* b1   = (const float*)d_in[8];
  const float* clsW = (const float*)d_in[9];
  const float* clsb = (const float*)d_in[10];
  const float* c_start = (const float*)d_in[11];
  const float* c_end   = (const float*)d_in[12];
  const float* c_trans = (const float*)d_in[13];
  float* out = (float*)d_out;  // out[0]=loss, out+1 = logits [B][T][L]

  char* ws = (char*)d_ws;
  size_t off = 0;
  auto take = [&](size_t bytes) -> void* {
    void* p = ws + off;
    off += (bytes + 255) & ~(size_t)255;
    return p;
  };
  float*  xg    = (float*) take((size_t)2 * kT * kB * k4H * 4); // 268 MB
  __bf16* xbf   = (__bf16*)take((size_t)kM * kD * 2);           //  50 MB
  __bf16* h0    = (__bf16*)take((size_t)kM * 512 * 2);          //  34 MB
  __bf16* h1    = (__bf16*)take((size_t)kM * 512 * 2);          //  34 MB
  __bf16* w0ihb = (__bf16*)take((size_t)2 * k4H * kD * 2);
  __bf16* w0hhb = (__bf16*)take((size_t)2 * k4H * kH * 2);
  __bf16* w1ihb = (__bf16*)take((size_t)2 * k4H * 512 * 2);
  __bf16* w1hhb = (__bf16*)take((size_t)2 * k4H * kH * 2);
  __bf16* clsWb = (__bf16*)take((size_t)16 * 512 * 2);

  auto cvt = [&](const float* src, __bf16* dst, int n) {
    f32_to_bf16_kernel<<<(n + 255) / 256, 256, 0, stream>>>(src, dst, n);
  };
  cvt(features, xbf, kM * kD);
  cvt(w0ih, w0ihb, 2 * k4H * kD);
  cvt(w0hh, w0hhb, 2 * k4H * kH);
  cvt(w1ih, w1ihb, 2 * k4H * 512);
  cvt(w1hh, w1hhb, 2 * k4H * kH);
  pad_cls_kernel<<<(16 * 512) / 256, 256, 0, stream>>>(clsW, clsWb);

  const size_t smemProj = 2 * 12 * 1024;                           //  24 KB
  const size_t smemScan = (size_t)kB * k4H * 4 + (size_t)kB * kH * 2;  // 288 KB

  // Layer 0: projections (both dirs across device), then persistent scans.
  proj_gemm_kernel<<<dim3(kM / 128, k4H / 64, 2), 256, smemProj, stream>>>(
      xbf, w0ihb, b0, xg, kD);
  lstm_scan_kernel<<<2, 1024, smemScan, stream>>>(xg, w0hhb, h0);
  // Layer 1
  proj_gemm_kernel<<<dim3(kM / 128, k4H / 64, 2), 256, smemProj, stream>>>(
      h0, w1ihb, b1, xg, 512);
  lstm_scan_kernel<<<2, 1024, smemScan, stream>>>(xg, w1hhb, h1);
  // Classifier + CRF
  cls_gemm_kernel<<<kM / 512, 256, 0, stream>>>(h1, clsWb, clsb, out + 1);
  crf_nll_kernel<<<1, 64, 0, stream>>>(out + 1, labels, c_start, c_end,
                                       c_trans, out);
}
